// Seq2Seq_77584289235014
// MI455X (gfx1250) — matrix operand order repaired and checked by reading, not compiled, not run on hardware
//
#include <hip/hip_runtime.h>

typedef _Float16 half_t;
typedef _Float16 v8h  __attribute__((ext_vector_type(8)));
typedef _Float16 v16h __attribute__((ext_vector_type(16)));
typedef float    v8f  __attribute__((ext_vector_type(8)));

#define Bz 64
#define Tz 1024
#define Hz 256
#define G3 768
#define AST 264   // LDS row stride (halves) for K=256 buffers: 4-bank stagger
#define DST 520   // LDS row stride (halves) for K=512 buffers: 4-bank stagger

// ---------------------------------------------------------------- helpers ---

__device__ __forceinline__ v8f wmma16(v16h a, v16h b, v8f c) {
  // D = A(16x32 f16) x B(32x16 f16) + C(16x16 f32)
  return __builtin_amdgcn_wmma_f32_16x16x32_f16(
      /*neg_a=*/false, a, /*neg_b=*/false, b,
      /*c_mod=*/(short)0, c, /*reuse_a=*/false, /*reuse_b=*/false);
}

// A fragment: lane holds row (lane&15), K chunks [base,base+8) and [base+16,base+24)
__device__ __forceinline__ v16h ldA(const half_t* p) {
  v8h lo = *(const v8h*)p;
  v8h hi = *(const v8h*)(p + 16);
  return __builtin_shufflevector(lo, hi, 0,1,2,3,4,5,6,7,8,9,10,11,12,13,14,15);
}
// B fragment: lane holds row (lane&15), K chunks [base,base+8) and [base+16,base+24)
__device__ __forceinline__ v16h ldB(const half_t* p) {
  v8h lo = *(const v8h*)p;
  v8h hi = *(const v8h*)(p + 16);
  return __builtin_shufflevector(lo, hi, 0,1,2,3,4,5,6,7,8,9,10,11,12,13,14,15);
}

__device__ __forceinline__ float fsig(float x)  { return 1.0f / (1.0f + __expf(-x)); }
__device__ __forceinline__ float ftanh(float x) { return 1.0f - 2.0f / (__expf(2.0f * x) + 1.0f); }

// ------------------------------------------------------------ prep kernels ---

__global__ void s2s_pack_f16(const float* __restrict__ src, half_t* __restrict__ dst, int n) {
  int i = blockIdx.x * blockDim.x + threadIdx.x;
  if (i < n) dst[i] = (half_t)src[i];
}

// dst row n (512 halves) = [Wih[n][0:256] | Whh[n][0:256]]
__global__ void s2s_pack_cat(const float* __restrict__ wih, const float* __restrict__ whh,
                             half_t* __restrict__ dst) {
  int i = blockIdx.x * blockDim.x + threadIdx.x;
  if (i < G3 * 512) {
    int row = i >> 9, k = i & 511;
    float v = (k < Hz) ? wih[row * Hz + k] : whh[row * Hz + (k - Hz)];
    dst[i] = (half_t)v;
  }
}

// X (B,T,2) -> Xt (T,B,2)
__global__ void s2s_transpose_x(const float* __restrict__ X, float* __restrict__ Xt) {
  int i = blockIdx.x * blockDim.x + threadIdx.x;
  if (i < Tz * Bz * 2) {
    int t = i / (Bz * 2), r = i % (Bz * 2), m = r >> 1, s = r & 1;
    Xt[i] = X[(size_t)m * Tz * 2 + t * 2 + s];
  }
}

// ----------------------------------------------- encoder GRU recurrence -----
// Single workgroup (8 wave32). Per step: gh = h @ Whh^T via WMMA (K=256,N=768),
// gx either inline (layer0, S=2) or loaded from precomputed buffer (layer1).
__global__ __launch_bounds__(256, 1) void s2s_gru_recur(
    const half_t* __restrict__ Wpk,     // (768,256) f16, row n = Whh[n][:]
    const float*  __restrict__ bhh,     // 768
    const float*  __restrict__ bih,     // 768 (layer0 only)
    const float*  __restrict__ Wih0,    // (768,2)  (layer0 only)
    const float*  __restrict__ Xt,      // (T,64,2) (layer0 only)
    const half_t* __restrict__ gxbuf,   // (T*64,768) f16 (layer1 only)
    half_t* __restrict__ y0out,         // (T*64,256) f16 (layer0 only)
    float*  __restrict__ stateOut)      // (64,256) final hidden
{
  __shared__ alignas(16) half_t lds[Bz * AST];   // h state, f16
  const int tid = threadIdx.x, lane = tid & 31, wv = tid >> 5;
  const int l15 = lane & 15, hih = lane >> 4;
  const int aoff = hih * 8, boff = hih * 8, mAdd = hih * 8;

  for (int i = tid; i < Bz * AST; i += 256) lds[i] = (half_t)0.f;

  const bool layer0 = (gxbuf == nullptr);
  float w0h[2][3] = {}, w1h[2][3] = {}, bi_h[2][3] = {}, bh_h[2][3] = {};
#pragma unroll
  for (int jt = 0; jt < 2; ++jt)
#pragma unroll
    for (int g = 0; g < 3; ++g) {
      const int jg = g * Hz + (2 * wv + jt) * 16 + l15;
      bh_h[jt][g] = bhh[jg];
      if (layer0) {
        w0h[jt][g] = Wih0[jg * 2];
        w1h[jt][g] = Wih0[jg * 2 + 1];
        bi_h[jt][g] = bih[jg];
      }
    }

  for (int t = 0; t < Tz; ++t) {
    __syncthreads();                       // h state ready
    if (!layer0 && (t + 1) < Tz)           // stream next step's gates into L2/L0
      __builtin_prefetch(gxbuf + (size_t)(t + 1) * Bz * G3 + tid * 192, 0, 1);

    v8f acc[2][3][4] = {};
    for (int kb = 0; kb < 8; ++kb) {
      const int k0 = kb * 32;
      v16h a[4];
#pragma unroll
      for (int mt = 0; mt < 4; ++mt)
        a[mt] = ldA(&lds[(mt * 16 + l15) * AST + k0 + aoff]);
#pragma unroll
      for (int jt = 0; jt < 2; ++jt)
#pragma unroll
        for (int g = 0; g < 3; ++g) {
          const int n = g * Hz + (2 * wv + jt) * 16 + l15;
          v16h b = ldB(Wpk + (size_t)n * Hz + k0 + boff);
#pragma unroll
          for (int mt = 0; mt < 4; ++mt)
            acc[jt][g][mt] = wmma16(a[mt], b, acc[jt][g][mt]);
        }
    }
    __syncthreads();                       // all GEMM reads of h done

#pragma unroll
    for (int jt = 0; jt < 2; ++jt) {
      const int j = (2 * wv + jt) * 16 + l15;
#pragma unroll
      for (int mt = 0; mt < 4; ++mt) {
#pragma unroll
        for (int v = 0; v < 8; ++v) {
          const int m = mt * 16 + v + mAdd;
          float gxr, gxz, gxn;
          if (layer0) {
            const float x0 = Xt[(size_t)t * Bz * 2 + m * 2 + 0];
            const float x1 = Xt[(size_t)t * Bz * 2 + m * 2 + 1];
            gxr = fmaf(x1, w1h[jt][0], fmaf(x0, w0h[jt][0], bi_h[jt][0]));
            gxz = fmaf(x1, w1h[jt][1], fmaf(x0, w0h[jt][1], bi_h[jt][1]));
            gxn = fmaf(x1, w1h[jt][2], fmaf(x0, w0h[jt][2], bi_h[jt][2]));
          } else {
            const half_t* gp = gxbuf + (size_t)(t * Bz + m) * G3;
            gxr = (float)gp[j];
            gxz = (float)gp[Hz + j];
            gxn = (float)gp[2 * Hz + j];
          }
          const float r  = fsig(gxr + acc[jt][0][mt][v] + bh_h[jt][0]);
          const float z  = fsig(gxz + acc[jt][1][mt][v] + bh_h[jt][1]);
          const float nn = ftanh(gxn + r * (acc[jt][2][mt][v] + bh_h[jt][2]));
          const float hp = (float)lds[m * AST + j];
          const float hn = (1.0f - z) * nn + z * hp;
          lds[m * AST + j] = (half_t)hn;
          if (y0out) y0out[(size_t)(t * Bz + m) * Hz + j] = (half_t)hn;
          if (t == Tz - 1) stateOut[m * Hz + j] = hn;
        }
      }
    }
  }
}

// -------------------------------- batched input-gate GEMM (encoder layer1) --
// gx1[r, :] = Y0[r, :] @ Wih1^T + bih1 ; r = t*64+m, 1024 blocks of 64 rows.
__global__ __launch_bounds__(256, 2) void s2s_xgates_gemm(
    const half_t* __restrict__ A,     // (T*64,256) f16
    const half_t* __restrict__ Wpk,   // (768,256) f16
    const float*  __restrict__ bih,   // 768
    half_t* __restrict__ out)         // (T*64,768) f16
{
  __shared__ alignas(16) half_t lds[Bz * AST];
  const int bt = blockIdx.x;
  const int tid = threadIdx.x, lane = tid & 31, wv = tid >> 5;
  const int l15 = lane & 15, hih = lane >> 4;
  const int aoff = hih * 8, boff = hih * 8, mAdd = hih * 8;

  for (int i = tid; i < Bz * 32; i += 256) {       // stage 64x256 halves
    const int row = i >> 5, c8 = i & 31;
    *(v8h*)&lds[row * AST + c8 * 8] = *(const v8h*)(A + (size_t)(bt * Bz + row) * Hz + c8 * 8);
  }
  __syncthreads();

  v8f acc[2][3][4] = {};
  for (int kb = 0; kb < 8; ++kb) {
    const int k0 = kb * 32;
    v16h a[4];
#pragma unroll
    for (int mt = 0; mt < 4; ++mt)
      a[mt] = ldA(&lds[(mt * 16 + l15) * AST + k0 + aoff]);
#pragma unroll
    for (int jt = 0; jt < 2; ++jt)
#pragma unroll
      for (int g = 0; g < 3; ++g) {
        const int n = g * Hz + (2 * wv + jt) * 16 + l15;
        v16h b = ldB(Wpk + (size_t)n * Hz + k0 + boff);
#pragma unroll
        for (int mt = 0; mt < 4; ++mt)
          acc[jt][g][mt] = wmma16(a[mt], b, acc[jt][g][mt]);
      }
  }

#pragma unroll
  for (int jt = 0; jt < 2; ++jt) {
    const int j = (2 * wv + jt) * 16 + l15;
#pragma unroll
    for (int g = 0; g < 3; ++g) {
      const float bb = bih[g * Hz + j];
#pragma unroll
      for (int mt = 0; mt < 4; ++mt)
#pragma unroll
        for (int v = 0; v < 8; ++v) {
          const int m = mt * 16 + v + mAdd;
          out[(size_t)(bt * Bz + m) * G3 + g * Hz + j] = (half_t)(acc[jt][g][mt][v] + bb);
        }
    }
  }
}

// ------------------------------------------------------------ decoder -------
struct DecBias { float brz[2][2]; float bnx[2]; float bnh[2]; };

// One fused GRU cell step: A = [x | h] (64x512 f16 in LDS), W row = [Wih|Whh].
// n-gate keeps x-path (kk<8) and h-path (kk>=8) in separate accumulators.
__device__ __forceinline__ void dec_layer(
    half_t* __restrict__ bufR,       // this layer's [x|h]; h slot updated
    half_t* __restrict__ bufO,       // other layer's buffer; x slot written
    const half_t* __restrict__ Wpk,  // (768,512) f16
    const DecBias& bb,
    half_t* __restrict__ out, int t,
    int lane, int wv)
{
  const int l15 = lane & 15, hih = lane >> 4;
  const int aoff = hih * 8, boff = hih * 8, mAdd = hih * 8;

  v8f arz[2][2][4] = {};
  v8f anx[2][4] = {};
  v8f anh[2][4] = {};
  for (int kk = 0; kk < 16; ++kk) {            // fused K = 512
    const int k0 = kk * 32;
    v16h a[4];
#pragma unroll
    for (int mt = 0; mt < 4; ++mt)
      a[mt] = ldA(&bufR[(mt * 16 + l15) * DST + k0 + aoff]);
#pragma unroll
    for (int jt = 0; jt < 2; ++jt) {
      const int jb = (2 * wv + jt) * 16 + l15;
#pragma unroll
      for (int g = 0; g < 2; ++g) {            // r,z: fused over x and h
        v16h b = ldB(Wpk + (size_t)(g * Hz + jb) * 512 + k0 + boff);
#pragma unroll
        for (int mt = 0; mt < 4; ++mt)
          arz[jt][g][mt] = wmma16(a[mt], b, arz[jt][g][mt]);
      }
      v16h bn = ldB(Wpk + (size_t)(2 * Hz + jb) * 512 + k0 + boff);
      if (kk < 8) {                            // uniform branch: EXEC unchanged
#pragma unroll
        for (int mt = 0; mt < 4; ++mt) anx[jt][mt] = wmma16(a[mt], bn, anx[jt][mt]);
      } else {
#pragma unroll
        for (int mt = 0; mt < 4; ++mt) anh[jt][mt] = wmma16(a[mt], bn, anh[jt][mt]);
      }
    }
  }
  __syncthreads();                             // all reads of bufR done

#pragma unroll
  for (int jt = 0; jt < 2; ++jt) {
    const int j = (2 * wv + jt) * 16 + l15;
#pragma unroll
    for (int mt = 0; mt < 4; ++mt)
#pragma unroll
      for (int v = 0; v < 8; ++v) {
        const int m = mt * 16 + v + mAdd;
        const float r  = fsig(arz[jt][0][mt][v] + bb.brz[jt][0]);
        const float z  = fsig(arz[jt][1][mt][v] + bb.brz[jt][1]);
        const float nn = ftanh(anx[jt][mt][v] + bb.bnx[jt] + r * (anh[jt][mt][v] + bb.bnh[jt]));
        const float hp = (float)bufR[m * DST + Hz + j];
        const float hn = (1.0f - z) * nn + z * hp;
        bufR[m * DST + Hz + j] = (half_t)hn;   // own h state
        bufO[m * DST + j]      = (half_t)hn;   // other layer's x input
        if (out) out[(size_t)(m * Tz + t) * Hz + j] = (half_t)hn;
      }
  }
  __syncthreads();
}

__global__ __launch_bounds__(256, 1) void s2s_decoder(
    const half_t* __restrict__ W0pk, const half_t* __restrict__ W1pk,  // (768,512)
    const float* __restrict__ dbih, const float* __restrict__ dbhh,    // (2,768)
    const float* __restrict__ hE0, const float* __restrict__ hE1,      // (64,256)
    half_t* __restrict__ dec_out)                                      // (64*1024,256)
{
  __shared__ alignas(16) half_t bufA0[Bz * DST];  // layer0: [x=h1_prev | h0]
  __shared__ alignas(16) half_t bufA1[Bz * DST];  // layer1: [x=h0n     | h1]
  const int tid = threadIdx.x, lane = tid & 31, wv = tid >> 5;
  const int l15 = lane & 15;

  for (int i = tid; i < Bz * Hz; i += 256) {
    const int m = i >> 8, j = i & 255;
    const float e0 = hE0[i], e1 = hE1[i];
    bufA0[m * DST + j]      = (half_t)ftanh(e1);  // first input = tanh(last enc out)
    bufA0[m * DST + Hz + j] = (half_t)e0;
    bufA1[m * DST + j]      = (half_t)0.f;
    bufA1[m * DST + Hz + j] = (half_t)e1;
  }

  DecBias b0, b1;
#pragma unroll
  for (int jt = 0; jt < 2; ++jt) {
    const int j = (2 * wv + jt) * 16 + l15;
    b0.brz[jt][0] = dbih[j] + dbhh[j];
    b0.brz[jt][1] = dbih[Hz + j] + dbhh[Hz + j];
    b0.bnx[jt] = dbih[2 * Hz + j];
    b0.bnh[jt] = dbhh[2 * Hz + j];
    b1.brz[jt][0] = dbih[G3 + j] + dbhh[G3 + j];
    b1.brz[jt][1] = dbih[G3 + Hz + j] + dbhh[G3 + Hz + j];
    b1.bnx[jt] = dbih[G3 + 2 * Hz + j];
    b1.bnh[jt] = dbhh[G3 + 2 * Hz + j];
  }
  __syncthreads();

  for (int t = 0; t < Tz; ++t) {
    dec_layer(bufA0, bufA1, W0pk, b0, nullptr, t, lane, wv);
    dec_layer(bufA1, bufA0, W1pk, b1, dec_out, t, lane, wv);
  }
}

// -------------------------------------------------------------- FC head -----
__global__ __launch_bounds__(256, 2) void s2s_fc(
    const half_t* __restrict__ dec,   // (65536,256) f16
    const half_t* __restrict__ W1pk,  // (128,256) f16
    const float* __restrict__ b1,     // 128
    const float* __restrict__ W2,     // (2,128) f32
    const float* __restrict__ b2,     // 2
    float* __restrict__ out)          // (65536,2)
{
  __shared__ alignas(16) half_t lds[Bz * AST];
  __shared__ float hid[Bz * 132];
  const int bt = blockIdx.x;
  const int tid = threadIdx.x, lane = tid & 31, wv = tid >> 5;
  const int l15 = lane & 15, hih = lane >> 4;
  const int aoff = hih * 8, boff = hih * 8, mAdd = hih * 8;

  for (int i = tid; i < Bz * 32; i += 256) {
    const int row = i >> 5, c8 = i & 31;
    *(v8h*)&lds[row * AST + c8 * 8] = *(const v8h*)(dec + (size_t)(bt * Bz + row) * Hz + c8 * 8);
  }
  __syncthreads();

  v8f acc[4] = {};                       // wave w owns hid cols [w*16, w*16+16)
  for (int kb = 0; kb < 8; ++kb) {
    const int k0 = kb * 32;
    v16h a[4];
#pragma unroll
    for (int mt = 0; mt < 4; ++mt)
      a[mt] = ldA(&lds[(mt * 16 + l15) * AST + k0 + aoff]);
    v16h b = ldB(W1pk + (size_t)(wv * 16 + l15) * Hz + k0 + boff);
#pragma unroll
    for (int mt = 0; mt < 4; ++mt)
      acc[mt] = wmma16(a[mt], b, acc[mt]);
  }

  const int col = wv * 16 + l15;
  const float bias1 = b1[col];
#pragma unroll
  for (int mt = 0; mt < 4; ++mt)
#pragma unroll
    for (int v = 0; v < 8; ++v) {
      const int m = mt * 16 + v + mAdd;
      float hv = acc[mt][v] + bias1;
      hv = (hv >= 0.0f) ? hv : 0.01f * hv;   // LeakyReLU
      hid[m * 132 + col] = hv;
    }
  __syncthreads();

  if (tid < 128) {                      // tiny 128->2 layer + tanh/sigmoid
    const int row = tid >> 1, s = tid & 1;
    const float* w2 = W2 + s * 128;
    float d = b2[s];
    for (int c = 0; c < 128; ++c) d = fmaf(hid[row * 132 + c], w2[c], d);
    out[(size_t)(bt * Bz + row) * 2 + s] = (s == 0) ? ftanh(d) : fsig(d);
  }
}

// ------------------------------------------------------------ launcher ------
extern "C" void kernel_launch(void* const* d_in, const int* in_sizes, int n_in,
                              void* d_out, int out_size, void* d_ws, size_t ws_size,
                              hipStream_t stream) {
  (void)in_sizes; (void)n_in; (void)out_size; (void)ws_size;
  const float* X     = (const float*)d_in[0];
  const float* eWih0 = (const float*)d_in[1];
  const float* eWih1 = (const float*)d_in[2];
  const float* eWhh  = (const float*)d_in[3];
  const float* eBih  = (const float*)d_in[4];
  const float* eBhh  = (const float*)d_in[5];
  const float* dWih  = (const float*)d_in[6];
  const float* dWhh  = (const float*)d_in[7];
  const float* dBih  = (const float*)d_in[8];
  const float* dBhh  = (const float*)d_in[9];
  const float* fcW1  = (const float*)d_in[10];
  const float* fcB1  = (const float*)d_in[11];
  const float* fcW2  = (const float*)d_in[12];
  const float* fcB2  = (const float*)d_in[13];
  float* out = (float*)d_out;

  char* ws = (char*)d_ws;
  size_t off = 0;
  auto alloc = [&](size_t bytes) -> void* {
    void* p = ws + off;
    off = (off + bytes + 255) & ~(size_t)255;
    return p;
  };
  half_t* Y0     = (half_t*)alloc((size_t)Tz * Bz * Hz * 2);   // enc L0 outputs
  half_t* GX1    = (half_t*)alloc((size_t)Tz * Bz * G3 * 2);   // enc L1 input gates
  half_t* DEC    = (half_t*)alloc((size_t)Bz * Tz * Hz * 2);   // decoder outputs
  float*  XT     = (float*) alloc((size_t)Tz * Bz * 2 * 4);
  float*  HE0    = (float*) alloc((size_t)Bz * Hz * 4);
  float*  HE1    = (float*) alloc((size_t)Bz * Hz * 4);
  half_t* eWhh0p = (half_t*)alloc((size_t)G3 * Hz * 2);
  half_t* eWhh1p = (half_t*)alloc((size_t)G3 * Hz * 2);
  half_t* eWih1p = (half_t*)alloc((size_t)G3 * Hz * 2);
  half_t* dW0p   = (half_t*)alloc((size_t)G3 * 512 * 2);
  half_t* dW1p   = (half_t*)alloc((size_t)G3 * 512 * 2);
  half_t* fcW1p  = (half_t*)alloc((size_t)128 * Hz * 2);

  // ---- weight packing (f32 -> f16, WMMA-friendly layouts) ----
  s2s_pack_f16<<<(G3 * Hz + 255) / 256, 256, 0, stream>>>(eWhh, eWhh0p, G3 * Hz);
  s2s_pack_f16<<<(G3 * Hz + 255) / 256, 256, 0, stream>>>(eWhh + (size_t)G3 * Hz, eWhh1p, G3 * Hz);
  s2s_pack_f16<<<(G3 * Hz + 255) / 256, 256, 0, stream>>>(eWih1, eWih1p, G3 * Hz);
  s2s_pack_f16<<<(128 * Hz + 255) / 256, 256, 0, stream>>>(fcW1, fcW1p, 128 * Hz);
  s2s_pack_cat<<<(G3 * 512 + 255) / 256, 256, 0, stream>>>(dWih, dWhh, dW0p);
  s2s_pack_cat<<<(G3 * 512 + 255) / 256, 256, 0, stream>>>(dWih + (size_t)G3 * Hz,
                                                           dWhh + (size_t)G3 * Hz, dW1p);
  s2s_transpose_x<<<(Tz * Bz * 2 + 255) / 256, 256, 0, stream>>>(X, XT);

  // ---- encoder layer 0 (sequential, single WGP, state in LDS) ----
  s2s_gru_recur<<<1, 256, 0, stream>>>(eWhh0p, eBhh, eBih, eWih0, XT,
                                       nullptr, Y0, HE0);
  // ---- encoder layer 1 input gates: one big batched WMMA GEMM ----
  s2s_xgates_gemm<<<Tz, 256, 0, stream>>>(Y0, eWih1p, eBih + G3, GX1);
  // ---- encoder layer 1 (sequential) ----
  s2s_gru_recur<<<1, 256, 0, stream>>>(eWhh1p, eBhh + G3, nullptr, nullptr, nullptr,
                                       GX1, nullptr, HE1);
  // ---- decoder: autoregressive, fused [x|h] K=512 GEMMs per layer ----
  s2s_decoder<<<1, 256, 0, stream>>>(dW0p, dW1p, dBih, dBhh, HE0, HE1, DEC);
  // ---- FC head (batched WMMA) + tanh/sigmoid normalization ----
  s2s_fc<<<Tz, 256, 0, stream>>>(DEC, fcW1p, fcB1, fcW2, fcB2, out);
}
